// GAT_56478819943002
// MI455X (gfx1250) — matrix operand order repaired and checked by reading, not compiled, not run on hardware
//
#include <hip/hip_runtime.h>
#include <math.h>

#define N_NODES 50000
#define N_EDGES 800000
#define NGRAPH  64

typedef __attribute__((ext_vector_type(16))) _Float16 v16h;
typedef __attribute__((ext_vector_type(8)))  float    v8f;

// -------- float atomic max via integer atomics (works for IEEE fp32) --------
__device__ __forceinline__ void atomic_max_f32(float* addr, float val) {
    if (val >= 0.0f) atomicMax((int*)addr, __float_as_int(val));
    else             atomicMin((unsigned int*)addr, __float_as_uint(val));
}

// ---------------- WMMA GEMM: C[M,N] = A[M,K] (fp32) x W[K,N] (fp32) ----------------
// Compile-time K and NN so every strided W load / C store gets an immediate offset.
// REQUIRES: M % 16 == 0 (true for every call site: 50000, 64).
// f16 inputs, fp32 accumulate, optional bias + activation epilogue.
// act: 0 = none, 1 = relu
template <int K, int NN>
__global__ __launch_bounds__(128)
void wmma_gemm_kernel(const float* __restrict__ A, const float* __restrict__ W,
                      const float* __restrict__ bias, float* __restrict__ C,
                      int M, int act) {
    constexpr int TILES_N = NN / 16;
    const int lane = threadIdx.x & 31;
    const int wave = threadIdx.x >> 5;
    const int t = blockIdx.x * 4 + wave;          // wave-uniform => EXEC stays full
    if (t >= (M >> 4) * TILES_N) return;
    const int tm = t / TILES_N;
    const int tn = t - tm * TILES_N;

    const int hi   = lane >> 4;                   // half-wave select
    const int mrow = tm * 16 + (lane & 15);       // A row this lane feeds
    const int ncol = tn * 16 + (lane & 15);       // B/C col this lane feeds
    const float* __restrict__ arow = A + (size_t)mrow * K;
    const float* __restrict__ wcol = W + ncol + (size_t)(8 * hi) * NN;

    v8f acc = {};
    #pragma unroll 4
    for (int k0 = 0; k0 < K; k0 += 32) {
        // A fragment (16x32 f16): two contiguous 8-float runs per lane:
        //   e=0..7  -> K = k0 + 8*hi + e ;  e=8..15 -> K = k0 + 16 + 8*hi + (e-8)
        const float* ap = arow + k0 + 8 * hi;
        float4 a0 = *(const float4*)(ap + 0);
        float4 a1 = *(const float4*)(ap + 4);
        float4 a2 = *(const float4*)(ap + 16);
        float4 a3 = *(const float4*)(ap + 20);
        if (k0 + 32 < K) __builtin_prefetch(arow + k0 + 32, 0, 1);
        v16h a;
        a[0]  = (_Float16)a0.x; a[1]  = (_Float16)a0.y;
        a[2]  = (_Float16)a0.z; a[3]  = (_Float16)a0.w;
        a[4]  = (_Float16)a1.x; a[5]  = (_Float16)a1.y;
        a[6]  = (_Float16)a1.z; a[7]  = (_Float16)a1.w;
        a[8]  = (_Float16)a2.x; a[9]  = (_Float16)a2.y;
        a[10] = (_Float16)a2.z; a[11] = (_Float16)a2.w;
        a[12] = (_Float16)a3.x; a[13] = (_Float16)a3.y;
        a[14] = (_Float16)a3.z; a[15] = (_Float16)a3.w;

        // B fragment (32x16 f16): e=0..7 -> K = k0 + 8*hi + e ; e=8..15 -> K = k0 + 16 + 8*hi + (e-8), fixed col ncol.
        // NN is constexpr => 16 loads with immediate offsets in one clause.
        const float* wp = wcol + (size_t)k0 * NN;
        v16h b;
        #pragma unroll
        for (int e = 0; e < 16; ++e) b[e] = (_Float16)wp[(e + (e >= 8 ? 8 : 0)) * NN];

        acc = __builtin_amdgcn_wmma_f32_16x16x32_f16(
                  false, a, false, b, (short)0, acc, false, false);
    }

    // C/D layout: VGPR v -> row = v + 8*hi, col = ncol
    float bval = bias ? bias[ncol] : 0.0f;
    float* cp = C + (size_t)(tm * 16 + 8 * hi) * NN + ncol;
    #pragma unroll
    for (int v = 0; v < 8; ++v) {
        float x = acc[v] + bval;
        if (act == 1) x = fmaxf(x, 0.0f);
        cp[v * NN] = x;
    }
}

static void launch_gemm(const float* A, const float* W, const float* bias, float* C,
                        int M, int K, int Nn, int act, hipStream_t s) {
    int tiles  = (M / 16) * (Nn / 16);
    int blocks = (tiles + 3) / 4;
#define GEMM_CASE(KK, NNN)                                                          \
    if (K == KK && Nn == NNN) {                                                     \
        wmma_gemm_kernel<KK, NNN><<<blocks, 128, 0, s>>>(A, W, bias, C, M, act);    \
        return;                                                                     \
    }
    GEMM_CASE(128, 32)
    GEMM_CASE(32, 64)
    GEMM_CASE(64, 128)
    GEMM_CASE(128, 1024)
    GEMM_CASE(1024, 512)
    GEMM_CASE(512, 128)
#undef GEMM_CASE
}

// ---------------- tiny tail GEMM (64x32 -> 4), negligible FLOPs ----------------
__global__ void small_gemm_kernel(const float* __restrict__ A, const float* __restrict__ W,
                                  const float* __restrict__ bias, float* __restrict__ C,
                                  int M, int K, int Nn) {
    int i = blockIdx.x * blockDim.x + threadIdx.x;
    if (i >= M * Nn) return;
    int r = i / Nn, c = i - r * Nn;
    float s = bias[c];
    const float* ar = A + (size_t)r * K;
    for (int k = 0; k < K; ++k) s += ar[k] * W[(size_t)k * Nn + c];
    C[i] = s;
}

// ---------------- fills / init ----------------
__global__ void fill_kernel(float* __restrict__ p, long n, float v) {
    long i = (long)blockIdx.x * blockDim.x + threadIdx.x;
    if (i < n) p[i] = v;
}
__global__ void bias_init_kernel(float* __restrict__ out, const float* __restrict__ b,
                                 long total, int d) {
    long i = (long)blockIdx.x * blockDim.x + threadIdx.x;
    if (i < total) out[i] = b[(int)(i & (d - 1))];   // d is a power of two (32/64/128)
}
__global__ void leaky_kernel(float* __restrict__ p, long n, float slope) {
    long i = (long)blockIdx.x * blockDim.x + threadIdx.x;
    if (i < n) { float v = p[i]; p[i] = (v > 0.0f) ? v : slope * v; }
}

// ---------------- edge phase ----------------
// e = a . leakyReLU(xl[src] + xr[dst], 0.2); segment-max into mx[dst]
__global__ void edge_logits_kernel(const float* __restrict__ xl, const float* __restrict__ xr,
                                   const float* __restrict__ a,
                                   const int* __restrict__ src, const int* __restrict__ dst,
                                   float* __restrict__ ev, float* __restrict__ mx,
                                   int E, int dout) {
    int e = blockIdx.x * blockDim.x + threadIdx.x;
    if (e >= E) return;
    int s = src[e], d = dst[e];
    const float* pl = xl + (size_t)s * dout;
    const float* pr = xr + (size_t)d * dout;
    float sum = 0.0f;
    for (int j = 0; j < dout; j += 4) {
        float4 l  = *(const float4*)(pl + j);
        float4 r  = *(const float4*)(pr + j);
        float4 aa = *(const float4*)(a + j);
        float t;
        t = l.x + r.x; sum += aa.x * ((t > 0.0f) ? t : 0.2f * t);
        t = l.y + r.y; sum += aa.y * ((t > 0.0f) ? t : 0.2f * t);
        t = l.z + r.z; sum += aa.z * ((t > 0.0f) ? t : 0.2f * t);
        t = l.w + r.w; sum += aa.w * ((t > 0.0f) ? t : 0.2f * t);
    }
    ev[e] = sum;
    atomic_max_f32(mx + d, sum);
}

// ex = exp(e - mx[dst]); segment-sum into dn[dst]
__global__ void edge_exp_kernel(const float* __restrict__ ev, const float* __restrict__ mx,
                                const int* __restrict__ dst,
                                float* __restrict__ ex, float* __restrict__ dn, int E) {
    int e = blockIdx.x * blockDim.x + threadIdx.x;
    if (e >= E) return;
    int d = dst[e];
    float v = __expf(ev[e] - mx[d]);
    ex[e] = v;
    unsafeAtomicAdd(dn + d, v);
}

// out[dst] += alpha * xl[src]   (out pre-seeded with bias)
__global__ void edge_aggr_kernel(const float* __restrict__ ex, const float* __restrict__ dn,
                                 const float* __restrict__ xl,
                                 const int* __restrict__ src, const int* __restrict__ dst,
                                 float* __restrict__ out, int E, int dout) {
    int e = blockIdx.x * blockDim.x + threadIdx.x;
    if (e >= E) return;
    int s = src[e], d = dst[e];
    float alpha = ex[e] / (dn[d] + 1e-16f);
    const float* p = xl + (size_t)s * dout;
    float*       o = out + (size_t)d * dout;
    for (int j = 0; j < dout; j += 4) {
        float4 v = *(const float4*)(p + j);
        unsafeAtomicAdd(o + j + 0, alpha * v.x);
        unsafeAtomicAdd(o + j + 1, alpha * v.y);
        unsafeAtomicAdd(o + j + 2, alpha * v.z);
        unsafeAtomicAdd(o + j + 3, alpha * v.w);
    }
}

// ---------------- global max pool (feature width fixed at 128) ----------------
__global__ void pool_kernel(const float* __restrict__ h, const int* __restrict__ batch,
                            float* __restrict__ g, int n) {
    long i = (long)blockIdx.x * blockDim.x + threadIdx.x;
    if (i >= (long)n * 128) return;
    int node = (int)(i >> 7);
    int j    = (int)(i & 127);
    atomic_max_f32(g + ((size_t)batch[node] << 7) + j, h[i]);
}

// ---------------- host orchestration ----------------
extern "C" void kernel_launch(void* const* d_in, const int* in_sizes, int n_in,
                              void* d_out, int out_size, void* d_ws, size_t ws_size,
                              hipStream_t stream) {
    const float* x     = (const float*)d_in[0];
    const int*   ei    = (const int*)d_in[1];
    const int*   src   = ei;
    const int*   dst   = ei + N_EDGES;
    const int*   batch = (const int*)d_in[2];
    const float* Wl[3] = {(const float*)d_in[3],  (const float*)d_in[7],  (const float*)d_in[11]};
    const float* Wr[3] = {(const float*)d_in[4],  (const float*)d_in[8],  (const float*)d_in[12]};
    const float* av[3] = {(const float*)d_in[5],  (const float*)d_in[9],  (const float*)d_in[13]};
    const float* bv[3] = {(const float*)d_in[6],  (const float*)d_in[10], (const float*)d_in[14]};
    const float* mw[5] = {(const float*)d_in[15], (const float*)d_in[17], (const float*)d_in[19],
                          (const float*)d_in[21], (const float*)d_in[23]};
    const float* mb[5] = {(const float*)d_in[16], (const float*)d_in[18], (const float*)d_in[20],
                          (const float*)d_in[22], (const float*)d_in[24]};
    const int din[3]  = {128, 32, 64};
    const int dout[3] = {32, 64, 128};

    // carve workspace
    char* wsp = (char*)d_ws;
    size_t off = 0;
    auto carve = [&](size_t bytes) -> float* {
        float* p = (float*)(wsp + off);
        off = (off + bytes + 255) & ~(size_t)255;
        return p;
    };
    const size_t nodeBuf = (size_t)N_NODES * 128 * sizeof(float);
    float* XL = carve(nodeBuf);
    float* XR = carve(nodeBuf);
    float* HA = carve(nodeBuf);
    float* HB = carve(nodeBuf);
    float* EV = carve((size_t)N_EDGES * sizeof(float));
    float* EX = carve((size_t)N_EDGES * sizeof(float));
    float* MX = carve((size_t)N_NODES * sizeof(float));
    float* DN = carve((size_t)N_NODES * sizeof(float));
    float* GP = carve((size_t)NGRAPH * 128 * sizeof(float));
    float* M1 = carve((size_t)NGRAPH * 1024 * sizeof(float));
    float* M2 = carve((size_t)NGRAPH * 512 * sizeof(float));
    float* M3 = carve((size_t)NGRAPH * 128 * sizeof(float));
    float* M4 = carve((size_t)NGRAPH * 32 * sizeof(float));
    (void)in_sizes; (void)n_in; (void)out_size; (void)ws_size;

    auto fill = [&](float* p, long n, float v) {
        fill_kernel<<<(int)((n + 255) / 256), 256, 0, stream>>>(p, n, v);
    };

    const int EB = (N_EDGES + 255) / 256;
    const float* hin = x;
    float* hout = HA;
    for (int L = 0; L < 3; ++L) {
        launch_gemm(hin, Wl[L], nullptr, XL, N_NODES, din[L], dout[L], 0, stream);
        launch_gemm(hin, Wr[L], nullptr, XR, N_NODES, din[L], dout[L], 0, stream);
        fill(MX, N_NODES, -INFINITY);
        fill(DN, N_NODES, 0.0f);
        edge_logits_kernel<<<EB, 256, 0, stream>>>(XL, XR, av[L], src, dst, EV, MX,
                                                   N_EDGES, dout[L]);
        edge_exp_kernel<<<EB, 256, 0, stream>>>(EV, MX, dst, EX, DN, N_EDGES);
        long total = (long)N_NODES * dout[L];
        bias_init_kernel<<<(int)((total + 255) / 256), 256, 0, stream>>>(hout, bv[L],
                                                                         total, dout[L]);
        edge_aggr_kernel<<<EB, 256, 0, stream>>>(EX, DN, XL, src, dst, hout,
                                                 N_EDGES, dout[L]);
        if (L < 2)
            leaky_kernel<<<(int)((total + 255) / 256), 256, 0, stream>>>(hout, total, 0.01f);
        hin = hout;
        hout = (hout == HA) ? HB : HA;
    }

    // global max pool per graph (hin = layer-3 output, d=128)
    fill(GP, (long)NGRAPH * 128, -INFINITY);
    long pn = (long)N_NODES * 128;
    pool_kernel<<<(int)((pn + 255) / 256), 256, 0, stream>>>(hin, batch, GP, N_NODES);

    // MLP head
    launch_gemm(GP, mw[0], mb[0], M1, NGRAPH, 128,  1024, 1, stream);
    launch_gemm(M1, mw[1], mb[1], M2, NGRAPH, 1024, 512,  1, stream);
    launch_gemm(M2, mw[2], mb[2], M3, NGRAPH, 512,  128,  1, stream);
    launch_gemm(M3, mw[3], mb[3], M4, NGRAPH, 128,  32,   1, stream);
    small_gemm_kernel<<<1, 256, 0, stream>>>(M4, mw[4], mb[4], (float*)d_out,
                                             NGRAPH, 32, 4);
}
